// MultiHeadAttention_13065290514659
// MI455X (gfx1250) — matrix operand order repaired and checked by reading, compiled, not run on hardware
//
#include <hip/hip_runtime.h>

// MultiHeadAttention forward for MI455X (gfx1250).
// wave32 + WMMA bf16 + async global->LDS staging (double-buffered, ASYNCcnt)
// + Tensor Data Mover for the strided V tiles (TENSORcnt).
// d_out = [ out: B*S*D f32 ][ attn: B*H*S*S f32 ]
// d_ws  = Qh | Kh | Vt | Ctx  (each B*S*D bf16; Vt is [B,H,DEPTH,S])

#define B_ 4
#define S_ 2048
#define D_ 1024
#define H_ 16
#define DEPTH_ 64
#define NT_ (S_ / 64)

typedef __attribute__((ext_vector_type(16))) __bf16 v16bf;
typedef __attribute__((ext_vector_type(8)))  float  v8f;
typedef __attribute__((ext_vector_type(4)))  unsigned int v4u;
typedef __attribute__((ext_vector_type(4)))  int v4i;
typedef __attribute__((ext_vector_type(8)))  int v8i;

union Frag16 { v16bf v; unsigned int u[8]; };

#if defined(__has_builtin)
#  if __has_builtin(__builtin_amdgcn_tensor_load_to_lds)
#    define HAVE_TDM 1
#  endif
#endif
#ifndef HAVE_TDM
#  define HAVE_TDM 0
#endif
// ROCm 7.2 clang-22 declares the 5-arg form; the toolchain that ships
// amd_gfx1250_TDM.h uses the 6-arg form.
#if defined(__has_include)
#  if __has_include(<hip/amd_detail/amd_gfx1250_TDM.h>)
#    define TDM_6ARG 1
#  endif
#endif
#ifndef TDM_6ARG
#  define TDM_6ARG 0
#endif

__device__ __forceinline__ unsigned short f32_to_bf16(float f) {
    unsigned int u = __float_as_uint(f);
    unsigned int r = 0x7FFFu + ((u >> 16) & 1u);   // round-to-nearest-even
    return (unsigned short)((u + r) >> 16);
}

__device__ __forceinline__ v8f zero8() {
    v8f z;
#pragma unroll
    for (int i = 0; i < 8; ++i) z[i] = 0.0f;
    return z;
}

__device__ __forceinline__ v8f wmma_bf16(const Frag16& a, const Frag16& b, v8f c) {
    return __builtin_amdgcn_wmma_f32_16x16x32_bf16(false, a.v, false, b.v,
                                                   (short)0, c, false, false);
}

// LDS byte offset of a __shared__ object (low 32 bits of the generic address).
__device__ __forceinline__ unsigned lds_off(const void* p) {
    return (unsigned)(unsigned long long)p;
}

// GLOBAL_LOAD_ASYNC_TO_LDS_B128: 16 bytes/lane, tracked by ASYNCcnt.
__device__ __forceinline__ void async_cp16(unsigned lds_byte, const void* g) {
    asm volatile("global_load_async_to_lds_b128 %0, %1, off"
                 :: "v"(lds_byte), "v"(g) : "memory");
}
#define ASYNC_WAIT(n) asm volatile("s_wait_asynccnt " #n ::: "memory")

__device__ __forceinline__ void tensor_wait0() {
#if defined(__has_builtin) && __has_builtin(__builtin_amdgcn_s_wait_tensorcnt)
    __builtin_amdgcn_s_wait_tensorcnt(0);
#else
    asm volatile("s_wait_tensorcnt 0" ::: "memory");
#endif
}

// One-shot TDM load of a 2D bf16 tile (tile_d0 x tile_d1 elements, row stride
// stride0 elements) from global into LDS.  Issue from a single wave.
__device__ __forceinline__ void tdm_load_2d(unsigned lds_byte, const void* gaddr,
                                            unsigned tensor_d0, unsigned tensor_d1,
                                            unsigned tile_d0, unsigned tile_d1,
                                            unsigned stride0) {
#if HAVE_TDM
    unsigned long long ga = (unsigned long long)gaddr;
    v4u g0;
    g0[0] = 1u;                                          // count=1 (valid), user mode
    g0[1] = lds_byte;                                    // lds_addr
    g0[2] = (unsigned)(ga & 0xFFFFFFFFu);                // global_addr[31:0]
    g0[3] = (unsigned)((ga >> 32) & 0x1FFFFFFu)          // global_addr[56:32]
          | (2u << 30);                                  // type=2 ("image")
    v8i g1;
    g1[0] = (int)(1u << 16);                             // data_size=1 -> 2 bytes
    g1[1] = (int)((tensor_d0 & 0xFFFFu) << 16);          // tensor_dim0[15:0]
    g1[2] = (int)((tensor_d0 >> 16) | ((tensor_d1 & 0xFFFFu) << 16));
    g1[3] = (int)((tensor_d1 >> 16) | (tile_d0 << 16));  // tile_dim0
    g1[4] = (int)(tile_d1 & 0xFFFFu);                    // tile_dim1 (tile_dim2=0)
    g1[5] = (int)stride0;                                // tensor_dim0_stride[31:0]
    g1[6] = 0;
    g1[7] = 0;
    v4i z4 = {0, 0, 0, 0};
#if TDM_6ARG
    v8i z8 = {0, 0, 0, 0, 0, 0, 0, 0};
    __builtin_amdgcn_tensor_load_to_lds(g0, g1, z4, z4, z8, 0);
#else
    __builtin_amdgcn_tensor_load_to_lds(g0, g1, z4, z4, 0);
#endif
#endif
}

// ---------------------------------------------------------------------------
// Tiled GEMM: out = A[M,K] * W[K,N] + bias[N].  Block 128x128, BK=32;
// 8 waves as 4x2, wave tile 32x64.
// A_BF16: 0 = A is f32, 1 = A is bf16.
// OUT_MODE: 0 = f32 row-major [M,N] (non-temporal)
//           1 = bf16 head-split  [B,H,S,DEPTH]
//           2 = bf16 head-split, V-transposed [B,H,DEPTH,S]
// ---------------------------------------------------------------------------
template <int A_BF16, int OUT_MODE>
__global__ void __launch_bounds__(256)
gemm_bias_kernel(const void* __restrict__ Ain, const float* __restrict__ W,
                 const float* __restrict__ bias, void* __restrict__ outv,
                 int M, int N, int K) {
    __shared__ unsigned short sA[128 * 32];   // [m][k]
    __shared__ unsigned short sB[128 * 32];   // transposed: [n][k]

    const int tid  = threadIdx.x;
    const int wave = tid >> 5;
    const int lane = tid & 31;
    const int g    = lane >> 4;
    const int l15  = lane & 15;
    const int wm   = wave >> 1;
    const int wn   = wave & 1;
    const int m0   = blockIdx.y * 128;
    const int n0   = blockIdx.x * 128;

    v8f acc[2][4];
#pragma unroll
    for (int mi = 0; mi < 2; ++mi)
#pragma unroll
        for (int ni = 0; ni < 4; ++ni) acc[mi][ni] = zero8();

    for (int k0 = 0; k0 < K; k0 += 32) {
        for (int i = tid; i < 128 * 32; i += 256) {
            int r = i >> 5, c = i & 31;
            if (A_BF16)
                sA[i] = ((const unsigned short*)Ain)[(size_t)(m0 + r) * K + k0 + c];
            else
                sA[i] = f32_to_bf16(((const float*)Ain)[(size_t)(m0 + r) * K + k0 + c]);
        }
        for (int i = tid; i < 128 * 32; i += 256) {
            int kk = i >> 7, n = i & 127;
            sB[n * 32 + kk] = f32_to_bf16(W[(size_t)(k0 + kk) * N + n0 + n]);
        }
        __syncthreads();

        Frag16 af[2], bf[4];
#pragma unroll
        for (int mi = 0; mi < 2; ++mi) {
            int m = wm * 32 + mi * 16 + l15;
#pragma unroll
            for (int p = 0; p < 8; ++p) {
                int kb = ((p >> 2) << 4) + (g << 3) + ((p & 3) << 1);
                af[mi].u[p] = *(const unsigned int*)(sA + m * 32 + kb);
            }
        }
#pragma unroll
        for (int ni = 0; ni < 4; ++ni) {
            int n = wn * 64 + ni * 16 + l15;
#pragma unroll
            for (int p = 0; p < 8; ++p) {
                int kb = ((p >> 2) << 4) + (g << 3) + ((p & 3) << 1);
                bf[ni].u[p] = *(const unsigned int*)(sB + n * 32 + kb);
            }
        }
#pragma unroll
        for (int mi = 0; mi < 2; ++mi)
#pragma unroll
            for (int ni = 0; ni < 4; ++ni)
                acc[mi][ni] = wmma_bf16(af[mi], bf[ni], acc[mi][ni]);
        __syncthreads();
    }

#pragma unroll
    for (int mi = 0; mi < 2; ++mi) {
#pragma unroll
        for (int ni = 0; ni < 4; ++ni) {
            int nn = n0 + wn * 64 + ni * 16 + l15;
            float bv = bias[nn];
#pragma unroll
            for (int r = 0; r < 8; ++r) {
                int mm = m0 + wm * 32 + mi * 16 + r + (g << 3);
                float v = acc[mi][ni][r] + bv;
                if (OUT_MODE == 0) {
                    __builtin_nontemporal_store(v, &((float*)outv)[(size_t)mm * N + nn]);
                } else {
                    int b = mm >> 11, s = mm & (S_ - 1);
                    int h = nn >> 6,  dp = nn & (DEPTH_ - 1);
                    if (OUT_MODE == 1)
                        ((unsigned short*)outv)[((size_t)(b * H_ + h) * S_ + s) * DEPTH_ + dp]
                            = f32_to_bf16(v);
                    else // OUT_MODE == 2: [B,H,DEPTH,S]
                        ((unsigned short*)outv)[((size_t)(b * H_ + h) * DEPTH_ + dp) * S_ + s]
                            = f32_to_bf16(v);
                }
            }
        }
    }
}

// ---------------------------------------------------------------------------
// Fused attention: per (b,h, 128 q-rows).  Three recompute passes over the
// 32 n-tiles of 64 columns:
//   pass A: logits (WMMA), row-max via shfl butterflies
//   pass B: logits, exp, row-sum via shfl butterflies
//   pass C: logits, p = exp*inv -> attn (non-temporal, written ONCE),
//           p->bf16 LDS, ctx += P @ V (WMMA), ctx -> bf16 [B,S,D]
// K tiles: double-buffered global_load_async_to_lds_b128 (prefetch nt+1
// while computing nt; in-order completion => s_wait_asynccnt 2).
// V tiles: one TDM tensor_load_to_lds per tile (wave 0), overlapped with the
// logits WMMA and published by the mid-iteration barrier.
// ---------------------------------------------------------------------------
__global__ void __launch_bounds__(256)
fused_attention_kernel(const unsigned short* __restrict__ Qh,
                       const unsigned short* __restrict__ Kh,
                       const unsigned short* __restrict__ Vt,
                       const float* __restrict__ mask,
                       float* __restrict__ attn,
                       unsigned short* __restrict__ Ctx) {
    __shared__ unsigned short sQ[128 * 64];     // [m][d]      16KB
    __shared__ unsigned short sK[2][64 * 64];   // [n][d] x2   16KB
    __shared__ unsigned short sV[64 * 64];      // [n][k]       8KB
    __shared__ unsigned short sP[128 * 64];     // [m][k]      16KB
    __shared__ float          sM[2][64];        // mask * -1e9 x2

    const int tid  = threadIdx.x;
    const int wave = tid >> 5;
    const int lane = tid & 31;
    const int g    = lane >> 4;
    const int l15  = lane & 15;

    const int qt = blockIdx.x & 15;
    const int bh = blockIdx.x >> 4;
    const int b  = bh >> 4;
    const int h  = bh & 15;

    const unsigned sQo = lds_off(sQ), sVo = lds_off(sV);
    const unsigned sKo[2] = { lds_off(sK[0]), lds_off(sK[1]) };
    const char* qg = (const char*)(Qh + ((size_t)bh * S_ + qt * 128) * DEPTH_);
    const char* kg = (const char*)(Kh + (size_t)bh * S_ * DEPTH_);
    const char* vg = (const char*)(Vt + (size_t)bh * DEPTH_ * S_);

    // K tile (8KB contiguous) + mask slice into buffer `buf`.
    auto stage_k = [&](int nt, int buf) {
#pragma unroll
        for (int it = 0; it < 2; ++it) {
            int c = tid + it * 256;
            async_cp16(sKo[buf] + c * 16, kg + (size_t)nt * 8192 + c * 16);
        }
        if (tid < 64) sM[buf][tid] = mask[(size_t)b * S_ + nt * 64 + tid] * (-1.0e9f);
    };

    // Logits for one 128x64 tile from sK[buf]; acc[ni] covers rows wave*16..+16.
    auto compute_logits = [&](v8f acc[4], int buf) {
        const unsigned short* sKb = sK[buf];
#pragma unroll
        for (int ni = 0; ni < 4; ++ni) acc[ni] = zero8();
#pragma unroll
        for (int st = 0; st < 2; ++st) {
            Frag16 af, bf[4];
            int m = wave * 16 + l15;
#pragma unroll
            for (int p = 0; p < 8; ++p) {
                int kb = st * 32 + ((p >> 2) << 4) + (g << 3) + ((p & 3) << 1);
                af.u[p] = *(const unsigned int*)(sQ + m * 64 + kb);
            }
#pragma unroll
            for (int ni = 0; ni < 4; ++ni) {
                int n = ni * 16 + l15;
#pragma unroll
                for (int p = 0; p < 8; ++p) {
                    int kb = st * 32 + ((p >> 2) << 4) + (g << 3) + ((p & 3) << 1);
                    bf[ni].u[p] = *(const unsigned int*)(sKb + n * 64 + kb);
                }
            }
#pragma unroll
            for (int ni = 0; ni < 4; ++ni)
                acc[ni] = wmma_bf16(af, bf[ni], acc[ni]);
        }
    };

    // Stage Q once (16KB contiguous) + first K tile; the in-order ASYNCcnt
    // window means the first s_wait_asynccnt 2 covers Q and K0.
#pragma unroll
    for (int it = 0; it < 4; ++it) {
        int c = tid + it * 256;
        async_cp16(sQo + c * 16, qg + (size_t)c * 16);
    }
    stage_k(0, 0);

    // ---- PASS A: row max --------------------------------------------------
    float rm[8];
#pragma unroll
    for (int r = 0; r < 8; ++r) rm[r] = -3.0e38f;

    for (int nt = 0; nt < NT_; ++nt) {
        int cur = nt & 1;
        if (nt + 1 < NT_) { stage_k(nt + 1, cur ^ 1); ASYNC_WAIT(2); }
        else              { ASYNC_WAIT(0); }
        __syncthreads();
        v8f acc[4];
        compute_logits(acc, cur);
#pragma unroll
        for (int r = 0; r < 8; ++r) {
            float t = -3.0e38f;
#pragma unroll
            for (int ni = 0; ni < 4; ++ni)
                t = fmaxf(t, acc[ni][r] * 0.125f + sM[cur][ni * 16 + l15]);
            t = fmaxf(t, __shfl_xor(t, 1));
            t = fmaxf(t, __shfl_xor(t, 2));
            t = fmaxf(t, __shfl_xor(t, 4));
            t = fmaxf(t, __shfl_xor(t, 8));
            rm[r] = fmaxf(rm[r], t);   // row m = wave*16 + 8*g + r
        }
        __syncthreads();
    }

    // ---- PASS B: row sum --------------------------------------------------
    float rs[8];
#pragma unroll
    for (int r = 0; r < 8; ++r) rs[r] = 0.0f;

    stage_k(0, 0);
    for (int nt = 0; nt < NT_; ++nt) {
        int cur = nt & 1;
        if (nt + 1 < NT_) { stage_k(nt + 1, cur ^ 1); ASYNC_WAIT(2); }
        else              { ASYNC_WAIT(0); }
        __syncthreads();
        v8f acc[4];
        compute_logits(acc, cur);
#pragma unroll
        for (int r = 0; r < 8; ++r) {
            float t = 0.0f;
#pragma unroll
            for (int ni = 0; ni < 4; ++ni)
                t += __expf(acc[ni][r] * 0.125f + sM[cur][ni * 16 + l15] - rm[r]);
            t += __shfl_xor(t, 1);
            t += __shfl_xor(t, 2);
            t += __shfl_xor(t, 4);
            t += __shfl_xor(t, 8);
            rs[r] += t;
        }
        __syncthreads();
    }
    float inv[8];
#pragma unroll
    for (int r = 0; r < 8; ++r) inv[r] = 1.0f / rs[r];

    // ---- PASS C: emit attn + ctx = P @ V ----------------------------------
    v8f cacc[4];
#pragma unroll
    for (int ni = 0; ni < 4; ++ni) cacc[ni] = zero8();

    stage_k(0, 0);
    for (int nt = 0; nt < NT_; ++nt) {
        int cur = nt & 1;
        // V tile [depth 64][seq 64] via TDM (one DMA, wave 0 only); the
        // previous iteration's trailing barrier guarantees sV is idle.
#if HAVE_TDM
        if (wave == 0)
            tdm_load_2d(sVo, vg + (size_t)nt * 128, S_, DEPTH_, 64, 64, S_);
#else
#pragma unroll
        for (int it = 0; it < 2; ++it) {
            int c = tid + it * 256;
            int n = c >> 3, w = c & 7;
            async_cp16(sVo + n * 128 + w * 16,
                       vg + (size_t)n * (S_ * 2) + nt * 128 + w * 16);
        }
#endif
        if (nt + 1 < NT_) { stage_k(nt + 1, cur ^ 1); ASYNC_WAIT(2); }
        else              { ASYNC_WAIT(0); }
        __syncthreads();

        v8f acc[4];
        compute_logits(acc, cur);

        // normalized p: stream to attn (final value, non-temporal), bf16 to LDS
#pragma unroll
        for (int ni = 0; ni < 4; ++ni) {
            int n = ni * 16 + l15;
#pragma unroll
            for (int r = 0; r < 8; ++r) {
                int m = wave * 16 + (g << 3) + r;
                float p = __expf(acc[ni][r] * 0.125f + sM[cur][n] - rm[r]) * inv[r];
                __builtin_nontemporal_store(
                    p, &attn[((size_t)bh * S_ + qt * 128 + m) * S_ + nt * 64 + n]);
                sP[m * 64 + n] = f32_to_bf16(p);
            }
        }
#if HAVE_TDM
        if (wave == 0) tensor_wait0();
#endif
        __syncthreads();   // publishes sP and sV

        // ctx += P(128x64) @ V(64x64)
#pragma unroll
        for (int st = 0; st < 2; ++st) {
            Frag16 af, bf[4];
            int m = wave * 16 + l15;
#pragma unroll
            for (int p = 0; p < 8; ++p) {
                int kb = st * 32 + ((p >> 2) << 4) + (g << 3) + ((p & 3) << 1);
                af.u[p] = *(const unsigned int*)(sP + m * 64 + kb);
            }
#pragma unroll
            for (int ni = 0; ni < 4; ++ni) {
                int n = ni * 16 + l15;
#pragma unroll
                for (int p = 0; p < 8; ++p) {
                    int kb = st * 32 + ((p >> 2) << 4) + (g << 3) + ((p & 3) << 1);
                    bf[ni].u[p] = *(const unsigned int*)(sV + n * 64 + kb);
                }
            }
#pragma unroll
            for (int ni = 0; ni < 4; ++ni)
                cacc[ni] = wmma_bf16(af, bf[ni], cacc[ni]);
        }
        __syncthreads();
    }

    // ctx epilogue -> bf16 [B, S, D] (heads re-merged)
#pragma unroll
    for (int ni = 0; ni < 4; ++ni) {
        int n = ni * 16 + l15;
#pragma unroll
        for (int r = 0; r < 8; ++r) {
            int m = qt * 128 + wave * 16 + (g << 3) + r;
            Ctx[((size_t)b * S_ + m) * D_ + h * DEPTH_ + n] = f32_to_bf16(cacc[ni][r]);
        }
    }
}

// ---------------------------------------------------------------------------
extern "C" void kernel_launch(void* const* d_in, const int* in_sizes, int n_in,
                              void* d_out, int out_size, void* d_ws, size_t ws_size,
                              hipStream_t stream) {
    const float* q    = (const float*)d_in[0];
    const float* k    = (const float*)d_in[1];
    const float* v    = (const float*)d_in[2];
    const float* mask = (const float*)d_in[3];
    const float* Wq   = (const float*)d_in[4];
    const float* bq   = (const float*)d_in[5];
    const float* Wk   = (const float*)d_in[6];
    const float* bk   = (const float*)d_in[7];
    const float* Wv   = (const float*)d_in[8];
    const float* bv   = (const float*)d_in[9];
    const float* Wo   = (const float*)d_in[10];
    const float* bo   = (const float*)d_in[11];

    float* out  = (float*)d_out;                         // [B,S,D]
    float* attn = out + (size_t)B_ * S_ * D_;            // [B,H,S,S]

    const size_t NH = (size_t)B_ * S_ * D_;              // elements per buffer
    unsigned short* wsh = (unsigned short*)d_ws;
    unsigned short* Qh  = wsh;                           // [B,H,S,DEPTH] bf16
    unsigned short* Kh  = Qh + NH;                       // [B,H,S,DEPTH] bf16
    unsigned short* Vt  = Kh + NH;                       // [B,H,DEPTH,S] bf16
    unsigned short* Ctx = Vt + NH;                       // [B,S,D]       bf16

    const int M = B_ * S_;                               // 8192
    dim3 gGemm(D_ / 128, M / 128);                       // 8 x 64

    // 1) QKV projections (f32 in -> bf16 head-split out)
    gemm_bias_kernel<0, 1><<<gGemm, 256, 0, stream>>>(q, Wq, bq, Qh, M, D_, D_);
    gemm_bias_kernel<0, 1><<<gGemm, 256, 0, stream>>>(k, Wk, bk, Kh, M, D_, D_);
    gemm_bias_kernel<0, 2><<<gGemm, 256, 0, stream>>>(v, Wv, bv, Vt, M, D_, D_);

    // 2) fused attention: attn written once; ctx in bf16 [B,S,D]
    fused_attention_kernel<<<B_ * H_ * (S_ / 128), 256, 0, stream>>>(
        Qh, Kh, Vt, mask, attn, Ctx);

    // 3) out = ctx @ Wo + bo (bf16 A, f32 out, non-temporal)
    gemm_bias_kernel<1, 0><<<gGemm, 256, 0, stream>>>(Ctx, Wo, bo, out, M, D_, D_);
}
